// Decoder_27419071218002
// MI455X (gfx1250) — matrix operand order repaired and checked by reading, not compiled, not run on hardware
//
#include <hip/hip_runtime.h>
#include <hip/hip_bf16.h>
#include <math.h>

// ---------------------------------------------------------------------------
// Pointer-network decoder for MI455X (gfx1250, wave32, WMMA).
//   B=128, S=1024, E=H=128.
//   ref = cc @ W_ref  (bf16 WMMA, f32 acc) -> 64MB, stays resident in 192MB L2
//   1023 sequential steps: q-proj (WMMA) -> logits (VALU tanh via v_exp_f32,
//   wave32 shfl reductions, Gumbel sampling, async-to-LDS q staging) ->
//   finalize (softmax/logp, mask+query update).
// ---------------------------------------------------------------------------

typedef __attribute__((ext_vector_type(16))) __bf16 v16bf;
typedef __attribute__((ext_vector_type(8)))  __bf16 v8bf;
typedef __attribute__((ext_vector_type(8)))  float  v8f;

#define DEC_B 128
#define DEC_S 1024
#define DEC_E 128
#define DEC_H 128
// Padded LDS stride for the transposed W tile (bf16 elems).
// 136*2 = 272B row = 68 dword-banks -> 4-bank rotation per column (n), and all
// fragment addresses (n*272 + kc*64 + {0,32}) are 16B aligned -> ds_load_b128.
#define LDW   136

__device__ __forceinline__ float fast_tanh(float x) {
    float ax = fabsf(x);
    float e  = __expf(-2.0f * ax);          // v_exp_f32
    float t  = (1.0f - e) / (1.0f + e);
    return copysignf(t, x);
}

__device__ __forceinline__ unsigned hash32(unsigned x) {
    x ^= x >> 16; x *= 0x7feb352dU;
    x ^= x >> 15; x *= 0x846ca68bU;
    x ^= x >> 16; return x;
}

// ---------------------------------------------------------------------------
// Generic 128-col GEMM tile kernel: OUT[row0:row0+128, 0:128] =
//   A[row0:row0+128, 0:128] @ W[128,128] (+bias).  row0 = blockIdx.x*128.
// 256 threads = 8 waves; wave w computes rows w*16..w*16+15 x all 128 cols.
// A fragments via float4 global loads (one-time 64MB stream for ref);
// W staged transposed in padded LDS; B fragments double-buffered across the
// 8 N-tiles (two ds_load_b128 each) so WMMAs don't stall on dscnt.
// ---------------------------------------------------------------------------
__global__ __launch_bounds__(256)
void gemm128_bf16_wmma(const float* __restrict__ A, const float* __restrict__ W,
                       const float* __restrict__ bias, float* __restrict__ OUT) {
    __shared__ __align__(16) __bf16 sWT[128 * LDW];   // sWT[n][k] = W[k][n]
    const int t = threadIdx.x;
    const size_t rowbase = (size_t)blockIdx.x * 128;

    #pragma unroll
    for (int it = 0; it < 64; ++it) {                 // stage 128x128 f32 -> bf16^T
        int idx = t + 256 * it;                       // idx = k*128 + n
        int k = idx >> 7, n = idx & 127;
        sWT[n * LDW + k] = (__bf16)W[idx];
    }
    __syncthreads();

    const int w  = t >> 5, l = t & 31;
    const int m0 = w * 16;
    const int arow = m0 + (l & 15);
    const int kbA = (l < 16) ? 0 : 8;                 // A-frag: lanes 0-15 K{0..7,16..23}
    const int kbB = (l < 16) ? 0 : 8;                 // B-frag: lanes 0-15 K{0..7,16..23}

    // Preload A fragments for the 4 K-chunks of 32 (K = 128), f32 -> bf16.
    v16bf afrag[4];
    #pragma unroll
    for (int kc = 0; kc < 4; ++kc) {
        const float4* pa4 =
            (const float4*)(A + (rowbase + arow) * 128 + kc * 32 + kbA);
        float4 x0 = pa4[0], x1 = pa4[1];              // K = kbA .. kbA+7
        float4 x2 = pa4[4], x3 = pa4[5];              // K = 16+kbA .. 16+kbA+7
        afrag[kc][0]  = (__bf16)x0.x; afrag[kc][1]  = (__bf16)x0.y;
        afrag[kc][2]  = (__bf16)x0.z; afrag[kc][3]  = (__bf16)x0.w;
        afrag[kc][4]  = (__bf16)x1.x; afrag[kc][5]  = (__bf16)x1.y;
        afrag[kc][6]  = (__bf16)x1.z; afrag[kc][7]  = (__bf16)x1.w;
        afrag[kc][8]  = (__bf16)x2.x; afrag[kc][9]  = (__bf16)x2.y;
        afrag[kc][10] = (__bf16)x2.z; afrag[kc][11] = (__bf16)x2.w;
        afrag[kc][12] = (__bf16)x3.x; afrag[kc][13] = (__bf16)x3.y;
        afrag[kc][14] = (__bf16)x3.z; afrag[kc][15] = (__bf16)x3.w;
    }

    auto loadB = [&](int nt, v16bf* dst) {
        const int n = nt * 16 + (l & 15);
        #pragma unroll
        for (int kc = 0; kc < 4; ++kc) {
            const __bf16* pb = &sWT[n * LDW + kc * 32 + kbB];   // 16B aligned
            v8bf lo = *(const v8bf*)pb;                          // ds_load_b128
            v8bf hi = *(const v8bf*)(pb + 16);                   // ds_load_b128
            dst[kc] = __builtin_shufflevector(lo, hi,
                0, 1, 2, 3, 4, 5, 6, 7, 8, 9, 10, 11, 12, 13, 14, 15);
        }
    };

    v16bf bbuf0[4], bbuf1[4];
    loadB(0, bbuf0);
    #pragma unroll
    for (int nt = 0; nt < 8; ++nt) {
        v16bf* bcur = (nt & 1) ? bbuf1 : bbuf0;
        v16bf* bnxt = (nt & 1) ? bbuf0 : bbuf1;
        if (nt < 7) loadB(nt + 1, bnxt);              // prefetch next N-tile's frags
        const int n = nt * 16 + (l & 15);
        v8f acc = {0.f, 0.f, 0.f, 0.f, 0.f, 0.f, 0.f, 0.f};
        #pragma unroll
        for (int kc = 0; kc < 4; ++kc)
            acc = __builtin_amdgcn_wmma_f32_16x16x32_bf16(
                false, afrag[kc], false, bcur[kc], (short)0, acc, false, false);
        const float bn = bias ? bias[n] : 0.0f;
        #pragma unroll
        for (int r = 0; r < 8; ++r) {                 // D: VGPR r -> M=r (+8 hi lanes)
            int grow = m0 + r + ((l >> 4) << 3);
            OUT[(rowbase + grow) * 128 + n] = acc[r] + bn;
        }
    }
}

// colmean[b,e] = mean_s cc[b,s,e]
__global__ __launch_bounds__(128)
void mean_kernel(const float* __restrict__ cc, float* __restrict__ colmean) {
    int b = blockIdx.x, e = threadIdx.x;
    const float* p = cc + (size_t)b * DEC_S * DEC_E + e;
    float s = 0.f;
    for (int i = 0; i < DEC_S; ++i) s += p[(size_t)i * DEC_E];
    colmean[b * DEC_E + e] = s * (1.0f / DEC_S);
}

// wvinit[e] = bv[e] + sum_j init_w[j] * Wv[j,e]   (j over 2E)
__global__ __launch_bounds__(128)
void wvinit_kernel(const float* __restrict__ init_w, const float* __restrict__ Wv,
                   const float* __restrict__ bv, float* __restrict__ wvinit) {
    int e = threadIdx.x;
    float a = bv[e];
    for (int j = 0; j < 2 * DEC_E; ++j) a += init_w[j] * Wv[j * DEC_E + e];
    wvinit[e] = a;
}

// query0 = hbar + wvinit (row broadcast)
__global__ __launch_bounds__(256)
void q0_kernel(const float* __restrict__ hbar, const float* __restrict__ wvinit,
               float* __restrict__ query) {
    int i = blockIdx.x * 256 + threadIdx.x;
    if (i < DEC_B * DEC_E) query[i] = hbar[i] + wvinit[i & (DEC_E - 1)];
}

// mask = high_mask with column 0 forced to 1
__global__ __launch_bounds__(256)
void mask_init_kernel(const float* __restrict__ high_mask, float* __restrict__ mask) {
    int i = blockIdx.x * 256 + threadIdx.x;
    if (i < DEC_B * DEC_S) {
        float m = high_mask[i];
        if ((i & (DEC_S - 1)) == 0) m = 1.0f;
        mask[i] = m;
    }
}

// ---------------------------------------------------------------------------
// Per-step logits: u[b,s] = 10*tanh( sum_h v[h]*tanh(ref[b,s,h]+q[b,h]) ),
// masked; per-(b,chunk) partial reduction: max logit + Gumbel argmax.
// grid (B, 8), 256 threads = 8 waves; each wave handles 16 s-values,
// 32 lanes split H=128 as 4 contiguous floats per lane (float4 loads).
// q[b,:] is staged to LDS via GLOBAL_LOAD_ASYNC_TO_LDS_B32 (ASYNCcnt).
// ---------------------------------------------------------------------------
__global__ __launch_bounds__(256)
void logits_kernel(int step, const float* __restrict__ ref, const float* __restrict__ q,
                   const float* __restrict__ v, const float* __restrict__ mask,
                   float* __restrict__ logits, float4* __restrict__ partials) {
    const int b = blockIdx.x, chunk = blockIdx.y;
    const int t = threadIdx.x, w = t >> 5, l = t & 31;

    __shared__ float sq[DEC_H];
    if (t < DEC_H) {   // waves 0-3 fully active -> EXEC all ones for async op
        // Flat LDS aperture: low 32 bits of a generic shared pointer are the
        // LDS byte offset (ISA 10.2 aperture mapping).
        unsigned lds_off = (unsigned)(size_t)&sq[t];
        const float* gp  = q + b * DEC_H + t;
        asm volatile("global_load_async_to_lds_b32 %0, %1, off"
                     :: "v"(lds_off), "v"(gp) : "memory");
        asm volatile("s_wait_asynccnt 0x0" ::: "memory");
    }
    __syncthreads();

    float vq[4], vv[4];
    #pragma unroll
    for (int j = 0; j < 4; ++j) { vq[j] = sq[4 * l + j]; vv[j] = v[4 * l + j]; }

    __shared__ float wmaxl[8], wmaxg[8];
    __shared__ int   warg[8];
    float bestl = -3e38f, bestg = -3e38f; int barg = 0;

    for (int it = 0; it < 16; ++it) {
        const int s = chunk * 128 + w * 16 + it;
        const float* rrow = ref + ((size_t)b * DEC_S + s) * DEC_H + 4 * l;
        if (s + 16 < DEC_S)
            __builtin_prefetch(rrow + 16 * DEC_H, 0, 1);   // global_prefetch_b8
        const float4 r4 = *(const float4*)rrow;
        float val = vv[0] * fast_tanh(r4.x + vq[0])
                  + vv[1] * fast_tanh(r4.y + vq[1])
                  + vv[2] * fast_tanh(r4.z + vq[2])
                  + vv[3] * fast_tanh(r4.w + vq[3]);
        #pragma unroll
        for (int m = 16; m > 0; m >>= 1) val += __shfl_xor(val, m, 32);
        if (l == 0) {
            float u  = 10.0f * fast_tanh(val);
            float lg = (mask[b * DEC_S + s] > 0.0f) ? -1e9f : u;
            logits[(size_t)b * DEC_S + s] = lg;
            unsigned h = hash32((unsigned)(step * 0x9E3779B9u)
                                ^ hash32((unsigned)b * 0x85EBCA6Bu ^ hash32((unsigned)s)));
            float uu = (float)(h >> 8) * (1.0f / 16777216.0f) + 2.9802322e-8f;
            float g  = -__logf(-__logf(uu));                // Gumbel(0,1)
            float cand = lg + g;
            bestl = fmaxf(bestl, lg);
            if (cand > bestg) { bestg = cand; barg = s; }
        }
    }
    if (l == 0) { wmaxl[w] = bestl; wmaxg[w] = bestg; warg[w] = barg; }
    __syncthreads();
    if (t == 0) {
        float M = -3e38f, G = -3e38f; int arg = 0;
        #pragma unroll
        for (int i = 0; i < 8; ++i) {
            M = fmaxf(M, wmaxl[i]);
            if (wmaxg[i] > G) { G = wmaxg[i]; arg = warg[i]; }
        }
        partials[b * 8 + chunk] = make_float4(M, G, __int_as_float(arg), 0.0f);
    }
}

// ---------------------------------------------------------------------------
// Per-step finalize (one block per b): global argmax/lse, logp, output,
// mask update, gather h, init_h freeze, query = hbar + bv + [init_h,h]@Wv.
// ---------------------------------------------------------------------------
__global__ __launch_bounds__(256)
void finalize_kernel(int step, const float* __restrict__ cc, const float* __restrict__ Wv,
                     const float* __restrict__ bv, const float* __restrict__ hbar,
                     const float* __restrict__ logits, const float4* __restrict__ partials,
                     float* __restrict__ mask, float* __restrict__ inith,
                     float* __restrict__ query, float* __restrict__ out) {
    const int b = blockIdx.x, t = threadIdx.x;
    __shared__ float sh_h[DEC_E], sh_ih[DEC_E], red[256];
    __shared__ float s_M;
    __shared__ int   s_idx;

    if (t == 0) {
        float M = -3e38f, G = -3e38f; int arg = 0;
        #pragma unroll
        for (int c = 0; c < 8; ++c) {
            float4 p = partials[b * 8 + c];
            M = fmaxf(M, p.x);
            if (p.y > G) { G = p.y; arg = __float_as_int(p.z); }
        }
        s_M = M; s_idx = arg;
    }
    __syncthreads();

    float sum = 0.f;
    for (int s = t; s < DEC_S; s += 256)
        sum += __expf(logits[(size_t)b * DEC_S + s] - s_M);
    red[t] = sum; __syncthreads();
    #pragma unroll
    for (int off = 128; off > 0; off >>= 1) {
        if (t < off) red[t] += red[t + off];
        __syncthreads();
    }
    const int idx = s_idx;
    if (t == 0) {
        float lse  = s_M + __logf(red[0]);
        float logp = logits[(size_t)b * DEC_S + idx] - lse;
        out[b * (DEC_S - 1) + step] = logp;                                   // logps.T
        out[DEC_B * (DEC_S - 1) + b * (DEC_S - 1) + step] = (float)idx;       // idxs.T
        mask[b * DEC_S + idx] = 1.0f;
    }
    __syncthreads();

    if (t < DEC_E) {
        float h = cc[((size_t)b * DEC_S + idx) * DEC_E + t];
        sh_h[t] = h;
        float ih = (step == 0) ? h : inith[b * DEC_E + t];
        sh_ih[t] = ih;
        if (step == 0) inith[b * DEC_E + t] = h;
    }
    __syncthreads();
    if (t < DEC_E) {
        float acc = hbar[b * DEC_E + t] + bv[t];
        #pragma unroll 4
        for (int j = 0; j < DEC_E; ++j) acc += sh_ih[j] * Wv[j * DEC_E + t];
        #pragma unroll 4
        for (int j = 0; j < DEC_E; ++j) acc += sh_h[j]  * Wv[(DEC_E + j) * DEC_E + t];
        query[b * DEC_E + t] = acc;
    }
}

// ---------------------------------------------------------------------------
// Host side. Inputs (setup_inputs order):
//  0 node_context 1 original_data 2 cell_context 3 high_mask 4 low_mask
//  5 init_w 6 Wc 7 bc 8 Wv 9 bv 10 W_ref 11 W_q 12 v
// ---------------------------------------------------------------------------
extern "C" void kernel_launch(void* const* d_in, const int* in_sizes, int n_in,
                              void* d_out, int out_size, void* d_ws, size_t ws_size,
                              hipStream_t stream) {
    (void)in_sizes; (void)n_in; (void)out_size; (void)ws_size;
    const float* cc     = (const float*)d_in[2];
    const float* hmask  = (const float*)d_in[3];
    const float* init_w = (const float*)d_in[5];
    const float* Wc     = (const float*)d_in[6];
    const float* bc     = (const float*)d_in[7];
    const float* Wv     = (const float*)d_in[8];
    const float* bv     = (const float*)d_in[9];
    const float* W_ref  = (const float*)d_in[10];
    const float* W_q    = (const float*)d_in[11];
    const float* v      = (const float*)d_in[12];
    float* out = (float*)d_out;

    // Workspace layout (floats). Total ~68 MB; ref (64MB) stays in 192MB L2.
    float* ws = (float*)d_ws;
    size_t o = 0;
    float* ref     = ws + o; o += (size_t)DEC_B * DEC_S * DEC_H;
    float* logits  = ws + o; o += (size_t)DEC_B * DEC_S;
    float* mask    = ws + o; o += (size_t)DEC_B * DEC_S;
    float* colmean = ws + o; o += (size_t)DEC_B * DEC_E;
    float* hbar    = ws + o; o += (size_t)DEC_B * DEC_E;
    float* wvinit  = ws + o; o += 128;
    float* query   = ws + o; o += (size_t)DEC_B * DEC_E;
    float* qbuf    = ws + o; o += (size_t)DEC_B * DEC_H;
    float* inith   = ws + o; o += (size_t)DEC_B * DEC_E;
    float4* partials = (float4*)(ws + o);

    // Setup
    mean_kernel<<<DEC_B, 128, 0, stream>>>(cc, colmean);
    wvinit_kernel<<<1, 128, 0, stream>>>(init_w, Wv, bv, wvinit);
    gemm128_bf16_wmma<<<1, 256, 0, stream>>>(colmean, Wc, bc, hbar);          // h_bar
    q0_kernel<<<(DEC_B * DEC_E + 255) / 256, 256, 0, stream>>>(hbar, wvinit, query);
    mask_init_kernel<<<(DEC_B * DEC_S + 255) / 256, 256, 0, stream>>>(hmask, mask);

    // ref = cc @ W_ref  (131072 x 128 x 128, bf16 WMMA)
    gemm128_bf16_wmma<<<(DEC_B * DEC_S) / 128, 256, 0, stream>>>(cc, W_ref, nullptr, ref);

    // Sequential decode: 1023 steps x 3 kernels (graph-capturable launch loop)
    for (int i = 0; i < DEC_S - 1; ++i) {
        gemm128_bf16_wmma<<<1, 256, 0, stream>>>(query, W_q, nullptr, qbuf);  // q = query@W_q
        logits_kernel<<<dim3(DEC_B, 8), 256, 0, stream>>>(i, ref, qbuf, v, mask,
                                                          logits, partials);
        finalize_kernel<<<DEC_B, 256, 0, stream>>>(i, cc, Wv, bv, hbar, logits,
                                                   partials, mask, inith, query, out);
    }
}
